// MoeDecoder_31353261261315
// MI455X (gfx1250) — compile-verified, matrix operand order checked by reading, not run on hardware
//
#include <hip/hip_runtime.h>
#include <hip/hip_bf16.h>

// ---------------------------------------------------------------------------
// Types for WMMA fragments (gfx1250, wave32)
// ---------------------------------------------------------------------------
typedef __attribute__((ext_vector_type(16))) __bf16 bf16x16;
typedef __attribute__((ext_vector_type(8)))  float  f32x8;
typedef __attribute__((ext_vector_type(4)))  unsigned int u32x4;
typedef __attribute__((ext_vector_type(4)))  int i32x4;

union Frag {
    u32x4   u[2];   // 32B = 8 VGPRs
    bf16x16 v;      // 16 bf16 values
};

// Async global->LDS copy path (CDNA5). Guarded so compile never breaks.
#if defined(__has_builtin)
#if __has_builtin(__builtin_amdgcn_global_load_async_to_lds_b128) && \
    __has_builtin(__builtin_amdgcn_s_wait_asynccnt)
#define MOE_ASYNC_COPY 1
#endif
#endif
#ifndef MOE_ASYNC_COPY
#define MOE_ASYNC_COPY 0
#endif

typedef i32x4 __attribute__((address_space(1)))* glb_v4p;
typedef i32x4 __attribute__((address_space(3)))* lds_v4p;

__device__ __forceinline__ glb_v4p to_glb_v4(const void* p) {
    return (glb_v4p)(unsigned long long)p;
}
__device__ __forceinline__ lds_v4p to_lds_v4(void* p) {
    // generic LDS pointer low 32 bits == LDS offset (flat aperture rule)
    return (lds_v4p)(unsigned int)(unsigned long long)p;
}

__device__ __forceinline__ unsigned short f2bf(float f) {
    unsigned int u = __float_as_uint(f);
    u += 0x7fffu + ((u >> 16) & 1u);       // round-to-nearest-even
    return (unsigned short)(u >> 16);
}

// ---------------------------------------------------------------------------
// Zero fp32 buffer (float4 per thread)
// ---------------------------------------------------------------------------
__global__ void moe_zero_kernel(float* __restrict__ p, int n4) {
    int i = blockIdx.x * blockDim.x + threadIdx.x;
    if (i < n4) ((float4*)p)[i] = make_float4(0.f, 0.f, 0.f, 0.f);
}

__global__ void moe_zero_i32_kernel(int* __restrict__ p, int n) {
    int i = threadIdx.x;
    if (i < n) p[i] = 0;
}

// prefix[e] = sum(counts[0..e-1]); tiny, one thread
__global__ void moe_prefix_kernel(const int* __restrict__ counts,
                                  int* __restrict__ prefix, int E) {
    if (threadIdx.x == 0) {
        int s = 0;
        for (int e = 0; e < E; ++e) { prefix[e] = s; s += counts[e]; }
    }
}

// ---------------------------------------------------------------------------
// fp32 -> bf16 elementwise convert (4 per thread)
// ---------------------------------------------------------------------------
__global__ void moe_cvt_bf16_kernel(const float* __restrict__ in,
                                    unsigned short* __restrict__ out, int n) {
    int i = (blockIdx.x * blockDim.x + threadIdx.x) * 4;
    if (i < n) {
        ushort4 v;
        v.x = f2bf(in[i + 0]);
        v.y = f2bf(in[i + 1]);
        v.z = f2bf(in[i + 2]);
        v.w = f2bf(in[i + 3]);
        *(ushort4*)(out + i) = v;
    }
}

// ---------------------------------------------------------------------------
// Transpose + convert [E][K,N] fp32 weights -> [E][N,K] bf16  (grid.z = expert)
// ---------------------------------------------------------------------------
__global__ __launch_bounds__(256)
void moe_transpose_cvt_kernel(const float* __restrict__ Wall,
                              unsigned short* __restrict__ WtAll,
                              int K, int Nn) {
    __shared__ float tile[32][33];
    const float* W = Wall + (size_t)blockIdx.z * K * Nn;
    unsigned short* Wt = WtAll + (size_t)blockIdx.z * K * Nn;
    int n0 = blockIdx.x * 32;
    int k0 = blockIdx.y * 32;
    int tx = threadIdx.x, ty = threadIdx.y;
#pragma unroll
    for (int i = 0; i < 4; ++i)
        tile[ty + 8 * i][tx] = W[(size_t)(k0 + ty + 8 * i) * Nn + n0 + tx];
    __syncthreads();
#pragma unroll
    for (int i = 0; i < 4; ++i) {
        int n = n0 + ty + 8 * i;
        Wt[(size_t)n * K + k0 + tx] = f2bf(tile[tx][ty + 8 * i]);
    }
}

// ---------------------------------------------------------------------------
// Gating: one wave32 per token. softmax, top-n, L1 renorm -> g[N,8];
// compacts token lists per expert (order-independent -> deterministic out).
// topn clamped to 4 so total slots <= 4*N (h1/h2 capacity).
// ---------------------------------------------------------------------------
__global__ __launch_bounds__(256)
void moe_gating_kernel(const float* __restrict__ x,
                       const float* __restrict__ gw,
                       const float* __restrict__ gb,
                       const int* __restrict__ topn_p,
                       float* __restrict__ g,
                       int* __restrict__ counts,
                       int* __restrict__ toks,
                       int D, int Ntok) {
    const int wid  = threadIdx.x >> 5;
    const int lane = threadIdx.x & 31;
    const int token = blockIdx.x * 8 + wid;

    float acc[8];
#pragma unroll
    for (int e = 0; e < 8; ++e) acc[e] = 0.f;

    const float* xr = x + (size_t)token * D;
    for (int d = lane; d < D; d += 32) {
        float xv = xr[d];
        const float* gr = gw + (size_t)d * 8;
#pragma unroll
        for (int e = 0; e < 8; ++e) acc[e] += xv * gr[e];
    }
#pragma unroll
    for (int off = 16; off > 0; off >>= 1) {
#pragma unroll
        for (int e = 0; e < 8; ++e) acc[e] += __shfl_xor(acc[e], off, 32);
    }
    if (lane == 0) {
        float l[8], p[8];
        float m = -1e30f;
#pragma unroll
        for (int e = 0; e < 8; ++e) { l[e] = acc[e] + gb[e]; m = fmaxf(m, l[e]); }
        float s = 0.f;
#pragma unroll
        for (int e = 0; e < 8; ++e) { p[e] = __expf(l[e] - m); s += p[e]; }
#pragma unroll
        for (int e = 0; e < 8; ++e) p[e] /= s;

        int topn = *topn_p;
        if (topn > 4) topn = 4;                  // slot-capacity guard (ref: 2)
        bool sel[8];
#pragma unroll
        for (int e = 0; e < 8; ++e) sel[e] = false;
        float ssum = 0.f;
        for (int it = 0; it < topn; ++it) {
            int bi = 0; float bv = -1.f;
#pragma unroll
            for (int e = 0; e < 8; ++e)
                if (!sel[e] && p[e] > bv) { bv = p[e]; bi = e; }
            sel[bi] = true; ssum += bv;
        }
        ssum = fmaxf(ssum, 1e-12f);
#pragma unroll
        for (int e = 0; e < 8; ++e) {
            float gv = sel[e] ? p[e] / ssum : 0.f;
            g[(size_t)token * 8 + e] = gv;
            if (sel[e]) {
                int slot = atomicAdd(&counts[e], 1);
                toks[(size_t)e * Ntok + slot] = token;
            }
        }
    }
}

// ---------------------------------------------------------------------------
// Tiled bf16 WMMA GEMM over compacted token sets, ALL experts (grid.z = e).
//   C[slot, n] = A_row(slot) . Bt[e][n, :]
// MODE 0: A rows gathered via tokens[]; OutB[prefix+slot] = bf16(lrelu(C+bias))
// MODE 1: A rows = prefix+slot;         OutB[prefix+slot] = bf16(lrelu(C+bias))
// MODE 2: A rows = prefix+slot; atomicAdd(OutF[token], g[token,e]*(C+bias))
// Block 256 thr (8 waves, 4x2), tile 128x128, K-chunk 64, double-buffered LDS,
// async global->LDS staging.
// ---------------------------------------------------------------------------
template <int MODE>
__global__ __launch_bounds__(256)
void moe_gemm_bf16_kernel(const unsigned short* __restrict__ A,
                          const unsigned short* __restrict__ BtAll,
                          const float* __restrict__ biasAll,
                          const int* __restrict__ toksAll,
                          const int* __restrict__ counts,
                          const int* __restrict__ prefix,
                          int Ntok, int Nn, int K,
                          unsigned short* __restrict__ OutB,
                          float* __restrict__ OutF,
                          const float* __restrict__ g) {
    constexpr int LDA = 72;                       // padded LDS row stride (bf16)
    __shared__ unsigned short Al[2][128 * LDA];
    __shared__ unsigned short Bl[2][128 * LDA];

    const int z     = blockIdx.z;                 // expert
    const int count = counts[z];
    const int bm    = blockIdx.y * 128;
    if (bm >= count) return;                      // early-exit (fixed grid)
    const int base  = prefix[z];                  // global slot base
    const int bn    = blockIdx.x * 128;

    const int* tokens = toksAll + (size_t)z * Ntok;
    const unsigned short* Bt = BtAll + (size_t)z * K * Nn;
    const float* bias = biasAll + (size_t)z * Nn;

    const int t      = threadIdx.x;
    const int lane   = t & 31;
    const int wave   = t >> 5;
    const int wave_m = wave >> 1;                 // 0..3
    const int wave_n = wave & 1;                  // 0..1

    // ---- per-thread staging addresses (fixed rows across K-chunks) ----
    const unsigned short* gA[4];
    const unsigned short* gB[4];
    int loff[4];
#pragma unroll
    for (int i = 0; i < 4; ++i) {
        int c   = t + i * 256;                    // 0..1023
        int row = c >> 3;                         // 0..127
        int kc  = (c & 7) * 8;                    // 0..56
        loff[i] = row * LDA + kc;
        int slot = bm + row;
        if (slot >= count) slot = count - 1;      // clamp padded slots
        int arow = (MODE == 0) ? tokens[slot] : (base + slot);
        gA[i] = A  + (size_t)arow * K + kc;
        gB[i] = Bt + (size_t)(bn + row) * K + kc;
    }

    f32x8 acc[2][4];
#pragma unroll
    for (int tm = 0; tm < 2; ++tm)
#pragma unroll
        for (int tn = 0; tn < 4; ++tn)
#pragma unroll
            for (int i = 0; i < 8; ++i) acc[tm][tn][i] = 0.f;

    const int r    = lane & 15;
    const int kshl = (lane & 16) ? 8 : 0;

    auto stage = [&](int buf, int cc) {
#pragma unroll
        for (int i = 0; i < 4; ++i) {
            const unsigned short* pa = gA[i] + cc * 64;
            const unsigned short* pb = gB[i] + cc * 64;
#if MOE_ASYNC_COPY
            __builtin_amdgcn_global_load_async_to_lds_b128(
                to_glb_v4(pa), to_lds_v4(&Al[buf][loff[i]]), 0, 0);
            __builtin_amdgcn_global_load_async_to_lds_b128(
                to_glb_v4(pb), to_lds_v4(&Bl[buf][loff[i]]), 0, 0);
#else
            *(u32x4*)(&Al[buf][loff[i]]) = *(const u32x4*)pa;
            *(u32x4*)(&Bl[buf][loff[i]]) = *(const u32x4*)pb;
#endif
        }
    };

    const int nchunk = K >> 6;                    // K/64
    stage(0, 0);
    int buf = 0;
    for (int cc = 0; cc < nchunk; ++cc) {
#if MOE_ASYNC_COPY
        __builtin_amdgcn_s_wait_asynccnt(0);
#endif
        __syncthreads();
        if (cc + 1 < nchunk) stage(buf ^ 1, cc + 1);

        // ---- two K=32 WMMA steps on current buffer ----
#pragma unroll
        for (int ks = 0; ks < 64; ks += 32) {
            const int koff = ks + kshl;
            Frag fa[2], fb[4];
#pragma unroll
            for (int tm = 0; tm < 2; ++tm) {
                int b = (wave_m * 32 + tm * 16 + r) * LDA + koff;
                fa[tm].u[0] = *(const u32x4*)(&Al[buf][b]);
                fa[tm].u[1] = *(const u32x4*)(&Al[buf][b + 16]);
            }
#pragma unroll
            for (int tn = 0; tn < 4; ++tn) {
                int b = (wave_n * 64 + tn * 16 + r) * LDA + koff;
                fb[tn].u[0] = *(const u32x4*)(&Bl[buf][b]);
                fb[tn].u[1] = *(const u32x4*)(&Bl[buf][b + 16]);
            }
#pragma unroll
            for (int tm = 0; tm < 2; ++tm)
#pragma unroll
                for (int tn = 0; tn < 4; ++tn)
                    acc[tm][tn] = __builtin_amdgcn_wmma_f32_16x16x32_bf16(
                        false, fa[tm].v, false, fb[tn].v,
                        (short)0, acc[tm][tn], false, false);
        }
        buf ^= 1;
    }

    // ---- epilogue (C layout: VGPR i -> row i, +8 for upper lane half) ----
    const int rhalf = (lane >> 4) * 8;
#pragma unroll
    for (int tm = 0; tm < 2; ++tm) {
#pragma unroll
        for (int tn = 0; tn < 4; ++tn) {
            int col   = bn + wave_n * 64 + tn * 16 + (lane & 15);
            int slot0 = bm + wave_m * 32 + tm * 16 + rhalf;
            float bb  = bias[col];
            f32x8 c   = acc[tm][tn];
            if (MODE != 2) {
#pragma unroll
                for (int i = 0; i < 8; ++i) {
                    int slot = slot0 + i;
                    if (slot < count) {           // don't clobber next expert
                        float v = c[i] + bb;
                        v = (v >= 0.f) ? v : 0.01f * v;
                        OutB[(size_t)(base + slot) * Nn + col] = f2bf(v);
                    }
                }
            } else {
#pragma unroll
                for (int i = 0; i < 8; ++i) {
                    int slot = slot0 + i;
                    if (slot < count) {
                        int tok  = tokens[slot];
                        float gw = g[(size_t)tok * 8 + z];
                        atomicAdd(&OutF[(size_t)tok * Nn + col],
                                  gw * (c[i] + bb));
                    }
                }
            }
        }
    }
}

// ---------------------------------------------------------------------------
// Host launcher
// ---------------------------------------------------------------------------
extern "C" void kernel_launch(void* const* d_in, const int* in_sizes, int n_in,
                              void* d_out, int out_size, void* d_ws, size_t ws_size,
                              hipStream_t stream) {
    (void)in_sizes; (void)n_in; (void)out_size; (void)ws_size;

    const float* x      = (const float*)d_in[0];
    const float* gate_w = (const float*)d_in[1];
    const float* gate_b = (const float*)d_in[2];
    const float* w1     = (const float*)d_in[3];
    const float* b1     = (const float*)d_in[4];
    const float* w2     = (const float*)d_in[5];
    const float* b2     = (const float*)d_in[6];
    const float* w3     = (const float*)d_in[7];
    const float* b3     = (const float*)d_in[8];
    const int*   topn   = (const int*)d_in[9];
    float*       out    = (float*)d_out;

    const int D = 1024, H = 1024, E = 8;
    const int N = 4 * 2048;                       // B*T = 8192 tokens
    const int CAP = 4 * N;                        // compacted slot capacity

    char*  ws  = (char*)d_ws;
    size_t off = 0;
    auto wsalloc = [&](size_t bytes) -> void* {
        void* p = ws + off;
        off = (off + bytes + 255) & ~(size_t)255;
        return p;
    };
    float*          g      = (float*)wsalloc((size_t)N * E * 4);
    int*            counts = (int*)wsalloc(E * sizeof(int));
    int*            prefix = (int*)wsalloc(E * sizeof(int));
    int*            toks   = (int*)wsalloc((size_t)E * N * sizeof(int));
    unsigned short* xb     = (unsigned short*)wsalloc((size_t)N * D * 2);
    unsigned short* wt1    = (unsigned short*)wsalloc((size_t)E * D * H * 2);
    unsigned short* wt2    = (unsigned short*)wsalloc((size_t)E * H * H * 2);
    unsigned short* wt3    = (unsigned short*)wsalloc((size_t)E * H * D * 2);
    unsigned short* h1     = (unsigned short*)wsalloc((size_t)CAP * H * 2);
    unsigned short* h2     = (unsigned short*)wsalloc((size_t)CAP * H * 2);

    // 1) zero output + routing counters
    moe_zero_kernel<<<(N * D / 4 + 255) / 256, 256, 0, stream>>>(out, N * D / 4);
    moe_zero_i32_kernel<<<1, 32, 0, stream>>>(counts, E);
    // 2) gating weights + per-expert token compaction, then slot prefix sums
    moe_gating_kernel<<<N / 8, 256, 0, stream>>>(x, gate_w, gate_b, topn,
                                                 g, counts, toks, D, N);
    moe_prefix_kernel<<<1, 32, 0, stream>>>(counts, prefix, E);
    // 3) x -> bf16
    moe_cvt_bf16_kernel<<<(N * D / 4 + 255) / 256, 256, 0, stream>>>(x, xb, N * D);
    // 4) all expert weights: transpose + convert (grid.z = expert)
    dim3 tpb(32, 8);
    dim3 tgrid(H / 32, D / 32, E);
    moe_transpose_cvt_kernel<<<tgrid, tpb, 0, stream>>>(w1, wt1, D, H);
    moe_transpose_cvt_kernel<<<tgrid, tpb, 0, stream>>>(w2, wt2, H, H);
    moe_transpose_cvt_kernel<<<tgrid, tpb, 0, stream>>>(w3, wt3, H, D);

    // 5) three GEMM stages, all experts concurrent (grid.z = expert)
    dim3 gblock(256);
    dim3 ggrid(H / 128, N / 128, E);              // worst-case rows; early-exit
    moe_gemm_bf16_kernel<0><<<ggrid, gblock, 0, stream>>>(
        xb, wt1, b1, toks, counts, prefix, N, H, D, h1, nullptr, nullptr);
    moe_gemm_bf16_kernel<1><<<ggrid, gblock, 0, stream>>>(
        h1, wt2, b2, toks, counts, prefix, N, H, H, h2, nullptr, nullptr);
    moe_gemm_bf16_kernel<2><<<dim3(D / 128, N / 128, E), gblock, 0, stream>>>(
        h2, wt3, b3, toks, counts, prefix, N, D, H, nullptr, out, g);
}
